// WindowAttention1D_41626823032870
// MI455X (gfx1250) — matrix operand order repaired and checked by reading, not compiled, not run on hardware
//
#include <hip/hip_runtime.h>
#include <hip/hip_bf16.h>

// ---- types -----------------------------------------------------------------
typedef __attribute__((ext_vector_type(16))) __bf16       v16bf;
typedef __attribute__((ext_vector_type(8)))  float        v8f;
typedef __attribute__((ext_vector_type(4)))  float        f32x4;
typedef __attribute__((ext_vector_type(4)))  unsigned int u32x4;
typedef __attribute__((ext_vector_type(2)))  unsigned int u32x2;

#define DIM   384
#define WIN   64
#define HEADS 12
#define HD    32

__device__ __forceinline__ v8f wmma_bf16(v16bf a, v16bf b, v8f c) {
  // D = A(16x32 bf16) * B(32x16 bf16) + C(16x16 f32)
  return __builtin_amdgcn_wmma_f32_16x16x32_bf16(false, a, false, b, (short)0, c,
                                                 false, false);
}

// A-fragment (16x32 bf16): lane holds row m=lane&15 (kgrp=lane>>4),
// 16 contiguous K values [kb..kb+15] -> two 16B loads.
__device__ __forceinline__ v16bf ld_afrag(const __bf16* p) {
  union { u32x4 u[2]; v16bf v; } r;
  r.u[0] = *(const u32x4*)(p);
  r.u[1] = *(const u32x4*)(p + 8);
  return r.v;
}

// B-fragment (32x16 bf16): lane holds col n=lane&15, 16 contiguous K values.
// Works for both LDS and global (bf16 weight copy) pointers.
__device__ __forceinline__ v16bf ld_bfrag(const __bf16* p) {
  union { u32x4 u[2]; v16bf v; } r;
  r.u[0] = *(const u32x4*)(p);
  r.u[1] = *(const u32x4*)(p + 8);
  return r.v;
}

__device__ __forceinline__ u32x2 pk4bf(f32x4 f) {
  union { __bf16 h[4]; u32x2 u; } pk;
  pk.h[0] = (__bf16)f.x; pk.h[1] = (__bf16)f.y;
  pk.h[2] = (__bf16)f.z; pk.h[3] = (__bf16)f.w;
  return pk.u;
}

// ---- one-time (per launch) weight conversion fp32 -> bf16 into workspace ---
#define QKV_F4  (1152 * DIM / 4)   // 110592 float4
#define PROJ_F4 (DIM * DIM / 4)    //  36864 float4
__global__ __launch_bounds__(256) void cvt_weights(
    const float* __restrict__ qkv_w, const float* __restrict__ proj_w,
    __bf16* __restrict__ qkv_wb, __bf16* __restrict__ proj_wb) {
  int idx = blockIdx.x * 256 + threadIdx.x;     // float4 granularity
  if (idx < QKV_F4) {
    ((u32x2*)qkv_wb)[idx] = pk4bf(((const f32x4*)qkv_w)[idx]);
  } else {
    int j = idx - QKV_F4;                        // j < PROJ_F4 by grid sizing
    ((u32x2*)proj_wb)[j] = pk4bf(((const f32x4*)proj_w)[j]);
  }
}

// ---- fused window-attention kernel: one workgroup (12 waves) per window ----
__global__ __launch_bounds__(384, 1) void wattn_fused(
    const float*  __restrict__ x,        // [B,64,384] fp32
    const __bf16* __restrict__ qkv_wb,   // [1152,384] bf16 (pre-converted)
    const __bf16* __restrict__ proj_wb,  // [384,384]  bf16 (pre-converted)
    const float*  __restrict__ proj_b,   // [384]
    const float*  __restrict__ bias_tab, // [127,12]
    float*        __restrict__ out)      // [B,64,384] fp32
{
  // LDS regions (bf16 unless noted), total 202752 B:
  //   [0      .. 49152)  xs : x tile 64x384      -> reused as O (os) later
  //   [49152  .. 98304)  qs : q (pre-scaled)     -> reused as P (ps) later
  //   [98304  ..147456)  ks : k                  -> reused as P (ps) later
  //   [147456..196608)  vst: v transposed 384x64
  //   [196608..202752)  bias table fp32 127x12
  __shared__ __align__(64) char smem[202752];
  __bf16* xs  = (__bf16*)(smem);
  __bf16* qs  = (__bf16*)(smem + 49152);
  __bf16* ks  = (__bf16*)(smem + 98304);
  __bf16* vst = (__bf16*)(smem + 147456);
  float*  bs  = (float*)(smem + 196608);

  const int tid  = threadIdx.x;
  const int lane = tid & 31;
  const int w    = tid >> 5;      // wave id (12 waves); also head id in phase 2
  const int hl   = lane >> 4;     // half-wave select
  const int cl   = lane & 15;     // column/row lane within a 16-wide tile
  const int b    = blockIdx.x;

  // ---- phase 0: x -> LDS (bf16), bias table -> LDS -------------------------
  const float* xg = x + (size_t)b * (WIN * DIM);
#pragma unroll
  for (int i = 0; i < 16; ++i) {
    int f4 = tid + i * 384;                       // 6144 float4 total
    *(u32x2*)(xs + 4 * f4) = pk4bf(*(const f32x4*)(xg + 4 * f4));
  }
  for (int i = tid; i < 127 * HEADS; i += 384) bs[i] = bias_tab[i];
  __syncthreads();

  // ---- phase 1: qkv = x @ qkv_w^T ------------------------------------------
  // Wave-uniform region split: waves 0-3 -> q, 4-7 -> k, 8-11 -> v.
  // Each wave owns 6 column tiles of its 384-wide region, in 2 blocks of 3.
  const float   scale  = 0.17677669529663687f;    // 32^-0.5 folded into q
  const int     reg    = w >> 2;                  // 0=q 1=k 2=v
  const int     wl     = w & 3;
  const __bf16* wbase  = qkv_wb + (size_t)(reg * DIM) * DIM;
  for (int ib = 0; ib < 2; ++ib) {
    v8f acc[3][4] = {};
    for (int k = 0; k < 12; ++k) {
      v16bf af[4];
#pragma unroll
      for (int mt = 0; mt < 4; ++mt)
        af[mt] = ld_afrag(xs + (mt * 16 + cl) * DIM + k * 32 + hl * 16);
#pragma unroll
      for (int i = 0; i < 3; ++i) {
        int   ntl = wl * 6 + ib * 3 + i;          // column tile in region
        v16bf bf  = ld_bfrag(wbase + (size_t)(ntl * 16 + cl) * DIM + k * 32 + hl * 16);
#pragma unroll
        for (int mt = 0; mt < 4; ++mt)
          acc[i][mt] = wmma_bf16(af[mt], bf, acc[i][mt]);
      }
    }
    // store: region branch is wave-uniform (no per-lane exec churn)
#pragma unroll
    for (int i = 0; i < 3; ++i) {
      int cbase = (wl * 6 + ib * 3 + i) * 16;
      if (reg == 0) {
#pragma unroll
        for (int mt = 0; mt < 4; ++mt)
#pragma unroll
          for (int r = 0; r < 8; ++r)
            qs[(mt * 16 + r + 8 * hl) * DIM + cbase + cl] =
                (__bf16)(acc[i][mt][r] * scale);
      } else if (reg == 1) {
#pragma unroll
        for (int mt = 0; mt < 4; ++mt)
#pragma unroll
          for (int r = 0; r < 8; ++r)
            ks[(mt * 16 + r + 8 * hl) * DIM + cbase + cl] = (__bf16)acc[i][mt][r];
      } else {
#pragma unroll
        for (int mt = 0; mt < 4; ++mt)
#pragma unroll
          for (int r = 0; r < 8; ++r)
            vst[(cbase + cl) * WIN + mt * 16 + r + 8 * hl] = (__bf16)acc[i][mt][r];
      }
    }
  }
  __syncthreads();

  // ---- phase 2a: S = q_h @ k_h^T + bias ; register softmax -----------------
  const int h = w;                                // one wave per head
  v8f S[4][4];
#pragma unroll
  for (int mt = 0; mt < 4; ++mt) {
    v16bf aq = ld_afrag(qs + (mt * 16 + cl) * DIM + h * HD + hl * 16);
#pragma unroll
    for (int nt = 0; nt < 4; ++nt) {
      v16bf bk = ld_bfrag(ks + (nt * 16 + cl) * DIM + h * HD + hl * 16);
      v8f z = {};
      S[mt][nt] = wmma_bf16(aq, bk, z);
    }
  }
#pragma unroll
  for (int mt = 0; mt < 4; ++mt) {
#pragma unroll
    for (int r = 0; r < 8; ++r) {
      const int row = mt * 16 + r + 8 * hl;       // query index
      float mx = -3.0e38f;
#pragma unroll
      for (int nt = 0; nt < 4; ++nt) {
        int   key = nt * 16 + cl;                 // key index
        float v   = S[mt][nt][r] + bs[(row - key + 63) * HEADS + h];
        S[mt][nt][r] = v;
        mx = fmaxf(mx, v);
      }
#pragma unroll
      for (int msk = 1; msk <= 8; msk <<= 1)      // reduce within 16-lane half
        mx = fmaxf(mx, __shfl_xor(mx, msk, 32));
      float sum = 0.f;
#pragma unroll
      for (int nt = 0; nt < 4; ++nt) {
        float e = __expf(S[mt][nt][r] - mx);
        S[mt][nt][r] = e;
        sum += e;
      }
#pragma unroll
      for (int msk = 1; msk <= 8; msk <<= 1)
        sum += __shfl_xor(sum, msk, 32);
      float inv = 1.0f / sum;
#pragma unroll
      for (int nt = 0; nt < 4; ++nt) S[mt][nt][r] *= inv;
    }
  }
  __syncthreads();  // all waves done reading qs/ks before P overlays them

  // ---- phase 2b: P -> LDS (overlay qs/ks); O = P @ v_h -> os (overlay xs) --
  __bf16* ps = (__bf16*)(smem + 49152) + h * (WIN * WIN);  // 8 KB per head
#pragma unroll
  for (int mt = 0; mt < 4; ++mt)
#pragma unroll
    for (int nt = 0; nt < 4; ++nt)
#pragma unroll
      for (int r = 0; r < 8; ++r)
        ps[(mt * 16 + r + 8 * hl) * WIN + nt * 16 + cl] = (__bf16)S[mt][nt][r];
  // same-wave LDS RAW: DScnt ordering + compiler waits make this safe
  __bf16* os = xs;  // x is dead; reuse region for attention output
#pragma unroll
  for (int mt = 0; mt < 4; ++mt)
#pragma unroll
    for (int o = 0; o < 2; ++o) {
      v8f acc = {};
#pragma unroll
      for (int c2 = 0; c2 < 2; ++c2) {            // K = 64 -> two k-steps
        v16bf ap = ld_afrag(ps + (mt * 16 + cl) * WIN + c2 * 32 + hl * 16);
        v16bf bv = ld_bfrag(vst + (h * HD + o * 16 + cl) * WIN + c2 * 32 + hl * 16);
        acc = wmma_bf16(ap, bv, acc);
      }
#pragma unroll
      for (int r = 0; r < 8; ++r)
        os[(mt * 16 + r + 8 * hl) * DIM + h * HD + o * 16 + cl] = (__bf16)acc[r];
    }
  __syncthreads();

  // ---- phase 3: out = O @ proj_w^T + proj_b --------------------------------
  float* og = out + (size_t)b * (WIN * DIM);
  {
    v8f acc[2][4] = {};
    for (int k = 0; k < 12; ++k) {
      v16bf af[4];
#pragma unroll
      for (int mt = 0; mt < 4; ++mt)
        af[mt] = ld_afrag(os + (mt * 16 + cl) * DIM + k * 32 + hl * 16);
#pragma unroll
      for (int i = 0; i < 2; ++i) {
        int   nt2 = w * 2 + i;                    // 24 col tiles / 12 waves
        v16bf bf  = ld_bfrag(proj_wb + (size_t)(nt2 * 16 + cl) * DIM + k * 32 + hl * 16);
#pragma unroll
        for (int mt = 0; mt < 4; ++mt)
          acc[i][mt] = wmma_bf16(af[mt], bf, acc[i][mt]);
      }
    }
#pragma unroll
    for (int i = 0; i < 2; ++i) {
      int   nt2 = w * 2 + i;
      float pb  = proj_b[nt2 * 16 + cl];
#pragma unroll
      for (int mt = 0; mt < 4; ++mt)
#pragma unroll
        for (int r = 0; r < 8; ++r)
          og[(size_t)(mt * 16 + r + 8 * hl) * DIM + nt2 * 16 + cl] =
              acc[i][mt][r] + pb;
    }
  }
}

// ---- harness entry ---------------------------------------------------------
extern "C" void kernel_launch(void* const* d_in, const int* in_sizes, int n_in,
                              void* d_out, int out_size, void* d_ws, size_t ws_size,
                              hipStream_t stream) {
  const float* x        = (const float*)d_in[0];
  const float* qkv_w    = (const float*)d_in[1];
  const float* proj_w   = (const float*)d_in[2];
  const float* proj_b   = (const float*)d_in[3];
  const float* bias_tab = (const float*)d_in[4];
  float*       out      = (float*)d_out;

  __bf16* qkv_wb  = (__bf16*)d_ws;                 //  884736 B
  __bf16* proj_wb = qkv_wb + 1152 * DIM;           // +294912 B (ws >= 1.13 MB)

  // 1) convert weights to bf16 once per launch (tiny: ~590K elements)
  cvt_weights<<<(QKV_F4 + PROJ_F4) / 256, 256, 0, stream>>>(qkv_w, proj_w,
                                                            qkv_wb, proj_wb);
  // 2) fused attention: one workgroup per window, 12 wave32 = 1 wave per head
  const int B = in_sizes[0] / (WIN * DIM);         // 4096
  wattn_fused<<<dim3(B), dim3(384), 0, stream>>>(x, qkv_wb, proj_wb, proj_b,
                                                 bias_tab, out);
}
